// SwitchAttention_86775519248809
// MI455X (gfx1250) — matrix operand order repaired and checked by reading, not compiled, not run on hardware
//
#include <hip/hip_runtime.h>
#include <hip/hip_bf16.h>
#include <math.h>

typedef __attribute__((ext_vector_type(16))) __bf16 v16bf;
typedef __attribute__((ext_vector_type(8)))  __bf16 v8bf;
typedef __attribute__((ext_vector_type(8)))  float  v8f;
typedef unsigned int u32x4 __attribute__((ext_vector_type(4)));
typedef int          i32x8 __attribute__((ext_vector_type(8)));
typedef int          i32x4 __attribute__((ext_vector_type(4)));

#ifndef __has_builtin
#define __has_builtin(x) 0
#endif
#if __has_builtin(__builtin_amdgcn_tensor_load_to_lds)
#define HAVE_TDM 1
#else
#define HAVE_TDM 0
#endif

#define WMMA_BF16(a, b, c) \
  __builtin_amdgcn_wmma_f32_16x16x32_bf16(false, (a), false, (b), (short)0, (c), false, false)

constexpr int BB  = 2;
constexpr int SS  = 2048;
constexpr int DD  = 1024;
constexpr int HH  = 16;
constexpr int DKK = 64;
constexpr int MT  = BB * SS;   // 4096 rows

// A-fragment loader for v_wmma_f32_16x16x32_bf16.
// Contiguous layout (matches B): lanes 0-15 hold K=0..15, lanes 16-31 hold
// K=16..31. Caller passes p = &A[row][kb + 16*(lane>>4)].
static __device__ inline v16bf ldA(const __bf16* p) {
  v8bf c0 = *(const v8bf*)(p);
  v8bf c1 = *(const v8bf*)(p + 8);
  return __builtin_shufflevector(c0, c1, 0, 1, 2, 3, 4, 5, 6, 7,
                                 8, 9, 10, 11, 12, 13, 14, 15);
}

// TDM 2-D tile load: D# per CDNA5 ISA ch.8. data_size=2B, count=1, type=2.
// lds_off is the workgroup-relative LDS byte offset; gaddr is the tile start.
static __device__ inline void tdm_load_2d(unsigned lds_off, unsigned long long gaddr,
                                          unsigned tensor_d0, unsigned tensor_d1,
                                          unsigned tile_d0, unsigned tile_d1,
                                          unsigned long long d0_stride) {
#if HAVE_TDM
  u32x4 g0;
  g0[0] = 1u;                                          // count=1, user mode
  g0[1] = lds_off;                                     // lds_addr
  g0[2] = (unsigned)(gaddr & 0xFFFFFFFFu);             // global_addr[31:0]
  g0[3] = (unsigned)((gaddr >> 32) & 0x01FFFFFFu) | (2u << 30);  // [56:32] | type=2
  i32x8 g1;
  g1[0] = (int)(1u << 16);                             // wg_mask=0, data_size=1 (2B)
  g1[1] = (int)((tensor_d0 & 0xFFFFu) << 16);          // tensor_dim0[15:0]
  g1[2] = (int)(((tensor_d0 >> 16) & 0xFFFFu) | ((tensor_d1 & 0xFFFFu) << 16));
  g1[3] = (int)(((tensor_d1 >> 16) & 0xFFFFu) | ((tile_d0 & 0xFFFFu) << 16));
  g1[4] = (int)(tile_d1 & 0xFFFFu);                    // tile_dim1 | tile_dim2=0
  g1[5] = (int)(d0_stride & 0xFFFFFFFFu);              // tensor_dim0_stride[31:0]
  g1[6] = (int)((d0_stride >> 32) & 0xFFFFu);          // stride[47:32] | dim1_stride=0
  g1[7] = 0;
  i32x4 gz4 = {0, 0, 0, 0};                            // groups 2/3 unused (2-D tile)
  i32x8 gz8 = {0, 0, 0, 0, 0, 0, 0, 0};
  __builtin_amdgcn_tensor_load_to_lds(g0, g1, gz4, gz4, gz8, 0);
#else
  (void)lds_off; (void)gaddr; (void)tensor_d0; (void)tensor_d1;
  (void)tile_d0; (void)tile_d1; (void)d0_stride;
#endif
}

static __device__ inline void wait_tensor0() {
#if HAVE_TDM
#if __has_builtin(__builtin_amdgcn_s_wait_tensorcnt)
  __builtin_amdgcn_s_wait_tensorcnt((short)0);
#else
  asm volatile("s_wait_tensorcnt 0" ::: "memory");
#endif
#endif
}

// ---------------------------------------------------------------------------
// Kernel 1: fp32 -> bf16 conversion; weights transposed to N-major (Wt[n][k]).
// ---------------------------------------------------------------------------
__global__ __launch_bounds__(256) void cvt_pack(
    const float* __restrict__ X, const float* __restrict__ wq,
    const float* __restrict__ wk, const float* __restrict__ wv,
    const float* __restrict__ wo, __bf16* __restrict__ Xb,
    __bf16* __restrict__ Wqt, __bf16* __restrict__ Wkt,
    __bf16* __restrict__ Wvt, __bf16* __restrict__ Wot) {
  size_t i = (size_t)blockIdx.x * 256 + threadIdx.x;
  const size_t nx = (size_t)MT * DD;                 // 4M elements of X
  if (i < nx) {
    Xb[i] = (__bf16)X[i];
    return;
  }
  size_t j = i - nx;                                 // 4 weight matrices, 1M each
  int w = (int)(j >> 20);
  size_t e = j & ((1u << 20) - 1);
  int n = (int)(e >> 10);
  int k = (int)(e & 1023);
  const float* src = (w == 0) ? wq : (w == 1) ? wk : (w == 2) ? wv : wo;
  __bf16* dst = (w == 0) ? Wqt : (w == 1) ? Wkt : (w == 2) ? Wvt : Wot;
  dst[e] = (__bf16)src[(size_t)k * DD + n];          // Wt[n][k] = W[k][n]
}

// ---------------------------------------------------------------------------
// Kernel 2: fused QKV projection. Wave -> one 32x32 tile of one of Q/K/V.
// Q,K stored [b][h][s][dk] bf16; V stored transposed [b][h][dk][s] bf16.
// ---------------------------------------------------------------------------
__global__ __launch_bounds__(256) void qkv_gemm(
    const __bf16* __restrict__ Xb, const __bf16* __restrict__ Wqt,
    const __bf16* __restrict__ Wkt, const __bf16* __restrict__ Wvt,
    __bf16* __restrict__ Q, __bf16* __restrict__ K, __bf16* __restrict__ Vt) {
  const int lane = threadIdx.x & 31;
  const int wid  = blockIdx.x * 8 + (threadIdx.x >> 5);
  const int mat  = wid / 4096;                       // 0=Q 1=K 2=V
  const int t    = wid % 4096;
  const int m0   = (t >> 5) * 32;
  const int n0   = (t & 31) * 32;
  const __bf16* Bt = (mat == 0) ? Wqt : (mat == 1) ? Wkt : Wvt;
  const int rsel = lane >> 4;
  const int rlo  = lane & 15;

  v8f acc[2][2] = {};
#pragma unroll 4
  for (int kb = 0; kb < DD; kb += 32) {
    const __bf16* pa = Xb + (size_t)(m0 + rlo) * DD + kb + 16 * rsel;
    v16bf a0 = ldA(pa);
    v16bf a1 = ldA(pa + (size_t)16 * DD);
    v16bf b0 = *(const v16bf*)(Bt + (size_t)(n0 + rlo) * DD + kb + 16 * rsel);
    v16bf b1 = *(const v16bf*)(Bt + (size_t)(n0 + 16 + rlo) * DD + kb + 16 * rsel);
    acc[0][0] = WMMA_BF16(a0, b0, acc[0][0]);
    acc[0][1] = WMMA_BF16(a0, b1, acc[0][1]);
    acc[1][0] = WMMA_BF16(a1, b0, acc[1][0]);
    acc[1][1] = WMMA_BF16(a1, b1, acc[1][1]);
  }

#pragma unroll
  for (int mi = 0; mi < 2; ++mi) {
#pragma unroll
    for (int ni = 0; ni < 2; ++ni) {
#pragma unroll
      for (int j = 0; j < 8; ++j) {
        int r = m0 + mi * 16 + j + 8 * rsel;         // C layout: M=j(+8), N=lane%16
        int c = n0 + ni * 16 + rlo;
        int bidx = r >> 11, s = r & (SS - 1);
        int h = c >> 6, dk = c & (DKK - 1);
        __bf16 v = (__bf16)acc[mi][ni][j];
        if (mat == 2)
          Vt[(((size_t)bidx * HH + h) * DKK + dk) * SS + s] = v;
        else if (mat == 0)
          Q[(((size_t)bidx * HH + h) * SS + s) * DKK + dk] = v;
        else
          K[(((size_t)bidx * HH + h) * SS + s) * DKK + dk] = v;
      }
    }
  }
}

// ---------------------------------------------------------------------------
// Kernel 3: flash attention with additive position bias.
// Block = 8 waves on one (b,h); K/V 32-key tiles staged in LDS by the TDM
// (double buffered, TENSORcnt), shared by all 8 waves (128 queries).
// ---------------------------------------------------------------------------
__global__ __launch_bounds__(256) void attn(
    const __bf16* __restrict__ Q, const __bf16* __restrict__ Kb,
    const __bf16* __restrict__ Vt, const float* __restrict__ bias,
    __bf16* __restrict__ ctx) {
  __shared__ __bf16 Kl[2][32 * DKK];                 // [key][d]   8 KB
  __shared__ __bf16 Vl[2][DKK * 32];                 // [d][key]   8 KB
  __shared__ __bf16 plds[8][16 * 32];                // per-wave P 8 KB
  const int lane  = threadIdx.x & 31;
  const int wslot = threadIdx.x >> 5;
  const int bh   = blockIdx.x >> 4;                  // 32 (b,h) pairs
  const int qblk = blockIdx.x & 15;                  // 16 blocks per (b,h)
  const int h = bh & (HH - 1);
  const int b = bh >> 4;
  const int q0 = (qblk * 8 + wslot) * 16;
  const int rsel = lane >> 4;
  const int rlo  = lane & 15;

  const __bf16* Qp = Q  + (((size_t)b * HH + h) * SS + q0) * DKK;
  const __bf16* Kp = Kb + ((size_t)b * HH + h) * SS * DKK;
  const __bf16* Vp = Vt + ((size_t)b * HH + h) * DKK * SS;
  const float*  bp = bias + ((size_t)h * SS + q0) * SS;
  __bf16* P = &plds[wslot][0];

  const __bf16* qb = Qp + (size_t)rlo * DKK + 16 * rsel;
  v16bf aq0 = ldA(qb);                               // d = 0..31
  v16bf aq1 = ldA(qb + 32);                          // d = 32..63

  v8f acc[4] = {};
  float mrow[8], lrow[8];
#pragma unroll
  for (int j = 0; j < 8; ++j) { mrow[j] = -INFINITY; lrow[j] = 0.f; }

  // ---- K/V tile staging (TDM preferred; sync copy fallback) ----
#if HAVE_TDM
#define STAGE(buf, kb)                                                         \
  do {                                                                         \
    if (wslot == 0) {                                                          \
      tdm_load_2d((unsigned)(uintptr_t)&Kl[(buf)][0],                          \
                  (unsigned long long)(uintptr_t)(Kp + (size_t)(kb) * DKK),    \
                  DKK, SS, DKK, 32, DKK);                                      \
      tdm_load_2d((unsigned)(uintptr_t)&Vl[(buf)][0],                          \
                  (unsigned long long)(uintptr_t)(Vp + (kb)),                  \
                  SS, DKK, 32, DKK, SS);                                       \
    }                                                                          \
  } while (0)
#define STAGE_FINISH()                                                         \
  do { if (wslot == 0) wait_tensor0(); } while (0)
#else
#define STAGE(buf, kb)                                                         \
  do {                                                                         \
    int t = threadIdx.x;                                                       \
    int kk = t >> 3, kc = t & 7;       /* K: 32 keys x 8 chunks of 8 elems */  \
    *(v8bf*)&Kl[(buf)][kk * DKK + kc * 8] =                                    \
        *(const v8bf*)(Kp + (size_t)((kb) + kk) * DKK + kc * 8);               \
    int dd = t >> 2, dc = t & 3;       /* V: 64 d x 4 chunks of 8 keys    */   \
    *(v8bf*)&Vl[(buf)][dd * 32 + dc * 8] =                                     \
        *(const v8bf*)(Vp + (size_t)dd * SS + (kb) + dc * 8);                  \
  } while (0)
#define STAGE_FINISH() do { } while (0)
#endif

  STAGE(0, 0);
  for (int it = 0; it < SS / 32; ++it) {
    const int kb = it * 32;
    STAGE_FINISH();
    __syncthreads();                                 // tile (it&1) ready; prev reads done
    if (it + 1 < SS / 32) STAGE((it + 1) & 1, kb + 32);

    const __bf16* KL = &Kl[it & 1][0];
    const __bf16* VL = &Vl[it & 1][0];

    float s0[8], s1[8];
    {  // score tile 0: keys kb..kb+15
      const __bf16* kp = KL + (size_t)rlo * DKK + 16 * rsel;
      v16bf bk0 = *(const v16bf*)(kp);
      v16bf bk1 = *(const v16bf*)(kp + 32);
      v8f sc = {};
      sc = WMMA_BF16(aq0, bk0, sc);
      sc = WMMA_BF16(aq1, bk1, sc);
#pragma unroll
      for (int j = 0; j < 8; ++j)
        s0[j] = sc[j] + bp[(size_t)(j + 8 * rsel) * SS + kb + rlo];
    }
    {  // score tile 1: keys kb+16..kb+31
      const __bf16* kp = KL + (size_t)(16 + rlo) * DKK + 16 * rsel;
      v16bf bk0 = *(const v16bf*)(kp);
      v16bf bk1 = *(const v16bf*)(kp + 32);
      v8f sc = {};
      sc = WMMA_BF16(aq0, bk0, sc);
      sc = WMMA_BF16(aq1, bk1, sc);
#pragma unroll
      for (int j = 0; j < 8; ++j)
        s1[j] = sc[j] + bp[(size_t)(j + 8 * rsel) * SS + kb + 16 + rlo];
    }

    // online softmax per row (row j+8*rsel lives in this lane's 16-lane group)
#pragma unroll
    for (int j = 0; j < 8; ++j) {
      float v = fmaxf(s0[j], s1[j]);
      v = fmaxf(v, __shfl_xor(v, 1));
      v = fmaxf(v, __shfl_xor(v, 2));
      v = fmaxf(v, __shfl_xor(v, 4));
      v = fmaxf(v, __shfl_xor(v, 8));
      float mnew  = fmaxf(mrow[j], v);
      float scale = __expf(mrow[j] - mnew);
      float p0 = __expf(s0[j] - mnew);
      float p1 = __expf(s1[j] - mnew);
      float ps = p0 + p1;
      ps += __shfl_xor(ps, 1);
      ps += __shfl_xor(ps, 2);
      ps += __shfl_xor(ps, 4);
      ps += __shfl_xor(ps, 8);
      lrow[j] = lrow[j] * scale + ps;
      mrow[j] = mnew;
      acc[0][j] *= scale; acc[1][j] *= scale;
      acc[2][j] *= scale; acc[3][j] *= scale;
      int r = j + 8 * rsel;
      P[r * 32 + rlo]      = (__bf16)p0;
      P[r * 32 + 16 + rlo] = (__bf16)p1;
    }
    asm volatile("s_wait_dscnt 0" ::: "memory");     // P stores visible to re-read

    v16bf ap = ldA(P + rlo * 32 + 16 * rsel);        // P as A-fragment (16x32)
#pragma unroll
    for (int n = 0; n < 4; ++n) {                    // 4 d-tiles of 16
      v16bf bv = *(const v16bf*)(VL + (size_t)(n * 16 + rlo) * 32 + 16 * rsel);
      acc[n] = WMMA_BF16(ap, bv, acc[n]);
    }
  }

  // epilogue: ctx[b*S+q][h*64+d] bf16, row-major for the output GEMM
#pragma unroll
  for (int j = 0; j < 8; ++j) {
    float inv = 1.f / lrow[j];
    int r = q0 + j + 8 * rsel;
#pragma unroll
    for (int n = 0; n < 4; ++n) {
      int c = h * DKK + n * 16 + rlo;
      ctx[(size_t)(b * SS + r) * DD + c] = (__bf16)(acc[n][j] * inv);
    }
  }
#undef STAGE
#undef STAGE_FINISH
}

// ---------------------------------------------------------------------------
// Kernel 4: output projection ctx(bf16) x Wo -> fp32 out.
// ---------------------------------------------------------------------------
__global__ __launch_bounds__(256) void out_gemm(
    const __bf16* __restrict__ ctx, const __bf16* __restrict__ Wot,
    float* __restrict__ out) {
  const int lane = threadIdx.x & 31;
  const int wid  = blockIdx.x * 8 + (threadIdx.x >> 5);
  const int m0   = (wid >> 5) * 32;
  const int n0   = (wid & 31) * 32;
  const int rsel = lane >> 4;
  const int rlo  = lane & 15;

  v8f acc[2][2] = {};
#pragma unroll 4
  for (int kb = 0; kb < DD; kb += 32) {
    const __bf16* pa = ctx + (size_t)(m0 + rlo) * DD + kb + 16 * rsel;
    v16bf a0 = ldA(pa);
    v16bf a1 = ldA(pa + (size_t)16 * DD);
    v16bf b0 = *(const v16bf*)(Wot + (size_t)(n0 + rlo) * DD + kb + 16 * rsel);
    v16bf b1 = *(const v16bf*)(Wot + (size_t)(n0 + 16 + rlo) * DD + kb + 16 * rsel);
    acc[0][0] = WMMA_BF16(a0, b0, acc[0][0]);
    acc[0][1] = WMMA_BF16(a0, b1, acc[0][1]);
    acc[1][0] = WMMA_BF16(a1, b0, acc[1][0]);
    acc[1][1] = WMMA_BF16(a1, b1, acc[1][1]);
  }

#pragma unroll
  for (int mi = 0; mi < 2; ++mi)
#pragma unroll
    for (int ni = 0; ni < 2; ++ni)
#pragma unroll
      for (int j = 0; j < 8; ++j) {
        int r = m0 + mi * 16 + j + 8 * rsel;
        int c = n0 + ni * 16 + rlo;
        out[(size_t)r * DD + c] = acc[mi][ni][j];
      }
}

// ---------------------------------------------------------------------------
extern "C" void kernel_launch(void* const* d_in, const int* in_sizes, int n_in,
                              void* d_out, int out_size, void* d_ws, size_t ws_size,
                              hipStream_t stream) {
  const float* X    = (const float*)d_in[0];   // hidden_states [2,2048,1024]
  const float* bias = (const float*)d_in[1];   // position_bias [1,16,2048,2048]
  // d_in[2] = key_value_states (unused: self-attention branch)
  const float* wq = (const float*)d_in[3];
  const float* wk = (const float*)d_in[4];
  const float* wv = (const float*)d_in[5];
  const float* wo = (const float*)d_in[6];
  float* out = (float*)d_out;

  char* ws = (char*)d_ws;
  const size_t MB = (size_t)1 << 20;
  __bf16* Xb  = (__bf16*)(ws + 0 * MB);        // 8 MB  [4096,1024]
  __bf16* Wqt = (__bf16*)(ws + 8 * MB);        // 2 MB  [1024,1024] N-major
  __bf16* Wkt = (__bf16*)(ws + 10 * MB);       // 2 MB
  __bf16* Wvt = (__bf16*)(ws + 12 * MB);       // 2 MB
  __bf16* Wot = (__bf16*)(ws + 14 * MB);       // 2 MB
  __bf16* Q   = (__bf16*)(ws + 16 * MB);       // 8 MB  [b][h][s][dk]
  __bf16* K   = (__bf16*)(ws + 24 * MB);       // 8 MB  [b][h][s][dk]
  __bf16* Vt  = (__bf16*)(ws + 32 * MB);       // 8 MB  [b][h][dk][s]
  __bf16* ctx = (__bf16*)(ws + 40 * MB);       // 8 MB  [4096,1024]

  // 1) conversions: 4M (X) + 4M (weights) elements
  cvt_pack<<<(8u << 20) / 256, 256, 0, stream>>>(X, wq, wk, wv, wo,
                                                 Xb, Wqt, Wkt, Wvt, Wot);
  // 2) QKV: 3 * 128*32 tiles = 12288 waves, 8 waves/block
  qkv_gemm<<<1536, 256, 0, stream>>>(Xb, Wqt, Wkt, Wvt, Q, K, Vt);
  // 3) attention: 32 (b,h) x 16 blocks, 8 waves/block
  attn<<<512, 256, 0, stream>>>(Q, K, Vt, bias, ctx);
  // 4) output projection: 128*32 tiles = 4096 waves
  out_gemm<<<512, 256, 0, stream>>>(ctx, Wot, out);
}
